// FreeConvNetwork_3753801416735
// MI455X (gfx1250) — matrix operand order repaired and checked by reading, not compiled, not run on hardware
//
#include <hip/hip_runtime.h>

typedef __attribute__((ext_vector_type(16))) __bf16        bf16x16;
typedef __attribute__((ext_vector_type(8)))  float         v8f;
typedef __attribute__((ext_vector_type(4)))  unsigned int  u32x4;
typedef __attribute__((ext_vector_type(8)))  unsigned int  u32x8;

// ---------------------------------------------------------------------------
// WMMA fragment layouts (wave32, v_wmma_f32_16x16x32_bf16):
// A 16x32 (MxK): lanes 0-15 -> M=lane, K={kb..kb+7, kb+16..kb+23}, kb = 0|8
// B 32x16 (KxN): lanes 0-15 -> N=lane, K={kb..kb+7, kb+16..kb+23}, kb = 0|8 (same as A)
// C/D 16x16:     VGPR r -> M = r + (lane<16 ? 0 : 8), N = lane&15
// ---------------------------------------------------------------------------
static __device__ inline bf16x16 pack_frag(u32x4 lo, u32x4 hi) {
  u32x8 w;
  w[0] = lo[0]; w[1] = lo[1]; w[2] = lo[2]; w[3] = lo[3];
  w[4] = hi[0]; w[5] = hi[1]; w[6] = hi[2]; w[7] = hi[3];
  return __builtin_bit_cast(bf16x16, w);
}

#define WMMA_BF16(A, B, C) \
  __builtin_amdgcn_wmma_f32_16x16x32_bf16(false, (A), false, (B), (short)0, (C), false, false)

// ---------------------------------------------------------------------------
// LC1: x[8192,1,28,28] -> relu -> h1 bf16 channels-last [B][13][13][16]
// K=9 per output: VALU (too skinny for WMMA; BW-bound on x). 32B vector store.
// ---------------------------------------------------------------------------
__global__ void __launch_bounds__(256) k_lc1(const float* __restrict__ x,
                                             const float* __restrict__ w1,
                                             const float* __restrict__ b1,
                                             __bf16* __restrict__ h1) {
  int idx = blockIdx.x * 256 + threadIdx.x;            // over B*169
  if (idx >= 8192 * 169) return;
  int b = idx / 169, loc = idx % 169;
  int ho = loc / 13, wo = loc % 13;
  const float* xp = x + b * 784 + (2 * ho) * 28 + (2 * wo);
  float p[9];
#pragma unroll
  for (int kh = 0; kh < 3; ++kh)
#pragma unroll
    for (int kw = 0; kw < 3; ++kw) p[kh * 3 + kw] = xp[kh * 28 + kw];
  bf16x16 vv;
#pragma unroll
  for (int o = 0; o < 16; ++o) {
    const float* wp = w1 + o * 1521 + loc * 9;         // [16][169][9]
    float acc = b1[o * 169 + loc];
#pragma unroll
    for (int k = 0; k < 9; ++k) acc = fmaf(p[k], wp[k], acc);
    vv[o] = (__bf16)fmaxf(acc, 0.0f);
  }
  *((bf16x16*)(h1 + b * 2704 + loc * 16)) = vv;        // 32B aligned vector store
}

// ---------------------------------------------------------------------------
// LC2: per-location GEMM [8192 x 144] * [144 x 32], 36 locations.
// K order: k = kk*16 + c (kk = kh*3+kw). A rows contiguous per 16-ch chunk.
// grid = (36, 64 batch tiles of 128); block = 256 (8 waves, M16 each)
// ---------------------------------------------------------------------------
__global__ void __launch_bounds__(256) k_lc2(const __bf16* __restrict__ h1,
                                             const float* __restrict__ w2,
                                             const float* __restrict__ b2,
                                             __bf16* __restrict__ h2) {
  // B tile swizzled to fragment order: [kstep(5)][ntile(2)][lane(32)][16]
  __shared__ alignas(32) __bf16 Bs[5 * 2 * 32 * 16];   // 10 KB
  const int loc = blockIdx.x;                          // 0..35
  const int ho = loc / 6, wo = loc % 6;
  for (int e = threadIdx.x; e < 5120; e += 256) {
    int i = e & 15, ln = (e >> 4) & 31, nt = (e >> 9) & 1, kst = e >> 10;
    int k = kst * 32 + ((i & 8) ? 16 : 0) + ((ln & 16) ? 8 : 0) + (i & 7);       // K index (padded 144->160)
    int o = nt * 16 + (ln & 15);
    __bf16 v = (__bf16)0.0f;
    if (k < 144) {
      int c = k & 15, kk = k >> 4;                     // k = kk*16 + c
      v = (__bf16)w2[o * 5184 + c * 324 + loc * 9 + kk];  // [32][16][36][9]
    }
    Bs[e] = v;
  }
  __syncthreads();

  const int lane = threadIdx.x & 31, wv = threadIdx.x >> 5;
  const int m0 = (blockIdx.y * 8 + wv) * 16;
  const int am = lane & 15, kb = (lane & 16) ? 8 : 0;
  const __bf16* abase = h1 + (m0 + am) * 2704 + kb;    // lane base, 16B aligned
  const int iy = 2 * ho, ix = 2 * wo;
  v8f acc0 = {}, acc1 = {};
#pragma unroll
  for (int ks = 0; ks < 5; ++ks) {
    // fragment halves = chunks 2ks, 2ks+1 (compile-time); chunk 9 is zero pad
    const int c0 = 2 * ks, c1 = 2 * ks + 1;
    const int p0 = (iy + c0 / 3) * 13 + ix + c0 % 3;   // pixel offsets (scalar)
    u32x4 lo = *(const u32x4*)(abase + p0 * 16);
    u32x4 hi = {0u, 0u, 0u, 0u};
    if (c1 < 9) {
      const int p1 = (iy + c1 / 3) * 13 + ix + c1 % 3;
      hi = *(const u32x4*)(abase + p1 * 16);
    }
    bf16x16 a = pack_frag(lo, hi);
    bf16x16 bf0 = *(const bf16x16*)(&Bs[((ks * 2 + 0) * 32 + lane) * 16]);
    bf16x16 bf1 = *(const bf16x16*)(&Bs[((ks * 2 + 1) * 32 + lane) * 16]);
    acc0 = WMMA_BF16(a, bf0, acc0);
    acc1 = WMMA_BF16(a, bf1, acc1);
  }
  // epilogue: h2 channels-last [B][6][6][32]
  const int n = lane & 15, rbase = (lane & 16) ? 8 : 0;
#pragma unroll
  for (int r = 0; r < 8; ++r) {
    int bb = m0 + rbase + r;
    h2[bb * 1152 + loc * 32 + n]      = (__bf16)fmaxf(acc0[r] + b2[n * 36 + loc], 0.0f);
    h2[bb * 1152 + loc * 32 + n + 16] = (__bf16)fmaxf(acc1[r] + b2[(n + 16) * 36 + loc], 0.0f);
  }
}

// ---------------------------------------------------------------------------
// LC3: per-location GEMM [8192 x 288] * [288 x 64], 16 locations, K=288=9*32.
// K order: k = kk*32 + c. Chunk kk == kstep (compile-time).
// ---------------------------------------------------------------------------
__global__ void __launch_bounds__(256) k_lc3(const __bf16* __restrict__ h2,
                                             const float* __restrict__ w3,
                                             const float* __restrict__ b3,
                                             __bf16* __restrict__ h3) {
  // B tile swizzled: [kstep(9)][ntile(4)][lane(32)][16]
  __shared__ alignas(32) __bf16 Bs[9 * 4 * 32 * 16];   // 36 KB
  const int loc = blockIdx.x;                          // 0..15
  const int ho = loc >> 2, wo = loc & 3;
  for (int e = threadIdx.x; e < 18432; e += 256) {
    int i = e & 15, ln = (e >> 4) & 31, nt = (e >> 9) & 3, kst = e >> 11;
    int k = kst * 32 + ((i & 8) ? 16 : 0) + ((ln & 16) ? 8 : 0) + (i & 7);
    int o = nt * 16 + (ln & 15);
    int c = k & 31, kk = k >> 5;                       // k = kk*32 + c
    Bs[e] = (__bf16)w3[o * 4608 + c * 144 + loc * 9 + kk];  // [64][32][16][9]
  }
  __syncthreads();

  const int lane = threadIdx.x & 31, wv = threadIdx.x >> 5;
  const int m0 = (blockIdx.y * 8 + wv) * 16;
  const int am = lane & 15, kb = (lane & 16) ? 8 : 0;
  const __bf16* abase = h2 + (m0 + am) * 1152 + kb;    // channels-last [B][6][6][32]
  v8f acc[4] = {{}, {}, {}, {}};
#pragma unroll
  for (int ks = 0; ks < 9; ++ks) {
    const int pix = (ho + ks / 3) * 6 + wo + ks % 3;   // chunk == ks, compile-time kh/kw
    u32x4 lo = *(const u32x4*)(abase + pix * 32);      // c = kb..kb+7
    u32x4 hi = *(const u32x4*)(abase + pix * 32 + 16); // c = kb+16..kb+23
    bf16x16 a = pack_frag(lo, hi);
#pragma unroll
    for (int t = 0; t < 4; ++t) {
      bf16x16 bf = *(const bf16x16*)(&Bs[((ks * 4 + t) * 32 + lane) * 16]);
      acc[t] = WMMA_BF16(a, bf, acc[t]);
    }
  }
  // epilogue: h3 flat [B][1024], feature = o*16 + loc (reference reshape order)
  const int n = lane & 15, rbase = (lane & 16) ? 8 : 0;
#pragma unroll
  for (int t = 0; t < 4; ++t) {
    int o = t * 16 + n;
    float bias = b3[o * 16 + loc];
#pragma unroll
    for (int r = 0; r < 8; ++r) {
      int bb = m0 + rbase + r;
      h3[bb * 1024 + o * 16 + loc] = (__bf16)(acc[t][r] + bias);   // no relu here
    }
  }
}

// ---------------------------------------------------------------------------
// FC1: [8192,1024] x [1024,512]^T + bias, relu. Tile M128 x N64, K-chunk 32.
// ---------------------------------------------------------------------------
__global__ void __launch_bounds__(256) k_fc1(const __bf16* __restrict__ h3,
                                             const float* __restrict__ fw1,
                                             const float* __restrict__ fb1,
                                             __bf16* __restrict__ h4) {
  // B chunk swizzled: [ntile(4)][lane(32)][16]
  __shared__ alignas(32) __bf16 Bs[4 * 32 * 16];       // 4 KB
  const int nBase = blockIdx.x * 64;
  const int mBase = blockIdx.y * 128;
  const int lane = threadIdx.x & 31, wv = threadIdx.x >> 5;
  const int am = lane & 15, kb = (lane & 16) ? 8 : 0;
  const __bf16* Arow = h3 + (mBase + wv * 16 + am) * 1024 + kb;
  v8f acc[4] = {{}, {}, {}, {}};
  for (int kc = 0; kc < 32; ++kc) {
    for (int e = threadIdx.x; e < 2048; e += 256) {
      int i = e & 15, ln = (e >> 4) & 31, nt = e >> 9;
      int kk = ((i & 8) ? 16 : 0) + ((ln & 16) ? 8 : 0) + (i & 7);
      int n = nBase + nt * 16 + (ln & 15);
      Bs[e] = (__bf16)fw1[n * 1024 + kc * 32 + kk];
    }
    __syncthreads();
    const __bf16* p = Arow + kc * 32;
    bf16x16 a = pack_frag(*(const u32x4*)(p), *(const u32x4*)(p + 16));
#pragma unroll
    for (int t = 0; t < 4; ++t) {
      bf16x16 bf = *(const bf16x16*)(&Bs[(t * 32 + lane) * 16]);
      acc[t] = WMMA_BF16(a, bf, acc[t]);
    }
    __syncthreads();
    if (kc + 1 < 32) {                                 // global_prefetch_b8 of streams
      __builtin_prefetch(Arow + (kc + 1) * 32, 0, 0);
      __builtin_prefetch(&fw1[(nBase + (threadIdx.x & 63)) * 1024 + (kc + 1) * 32], 0, 0);
    }
  }
  const int n = lane & 15, rbase = (lane & 16) ? 8 : 0;
#pragma unroll
  for (int t = 0; t < 4; ++t) {
    int nn = nBase + t * 16 + n;
    float bias = fb1[nn];
#pragma unroll
    for (int r = 0; r < 8; ++r) {
      int m = mBase + wv * 16 + rbase + r;
      h4[m * 512 + nn] = (__bf16)fmaxf(acc[t][r] + bias, 0.0f);
    }
  }
}

// ---------------------------------------------------------------------------
// FC2: [8192,512] x [512,10]^T + bias -> f32 out. N=10: VALU with fw2 in LDS.
// fw2 staging is a pure 20KB f32 copy -> async Global->LDS DMA (ASYNCcnt path).
// ---------------------------------------------------------------------------
__global__ void __launch_bounds__(256) k_fc2(const __bf16* __restrict__ h4,
                                             const float* __restrict__ fw2,
                                             const float* __restrict__ fb2,
                                             float* __restrict__ out) {
  __shared__ alignas(16) float w[10 * 512];            // 20 KB
  {
    // 1280 x b128 transfers, 5 per lane: LDS[off] = fw2_bytes[off]
    const unsigned ldsBase = (unsigned)(unsigned long long)(const void*)w;
    const unsigned byteOff = threadIdx.x * 16u;
#pragma unroll
    for (int it = 0; it < 5; ++it) {
      unsigned off = (unsigned)it * 4096u + byteOff;
      asm volatile("global_load_async_to_lds_b128 %0, %1, %2"
                   :: "v"(ldsBase + off), "v"(off), "s"(fw2)
                   : "memory");
    }
    asm volatile("s_wait_asynccnt 0x0" ::: "memory");  // this wave's DMAs done
  }
  __syncthreads();                                      // all waves' DMAs done
  int t = blockIdx.x * 256 + threadIdx.x;              // 81920 total
  int b = t / 10, j = t % 10;
  float acc = fb2[j];
  const __bf16* hp = h4 + b * 512;
  const float* wp = w + j * 512;
#pragma unroll 8
  for (int k = 0; k < 512; ++k) acc = fmaf((float)hp[k], wp[k], acc);
  out[t] = acc;
}

// ---------------------------------------------------------------------------
extern "C" void kernel_launch(void* const* d_in, const int* in_sizes, int n_in,
                              void* d_out, int out_size, void* d_ws, size_t ws_size,
                              hipStream_t stream) {
  (void)in_sizes; (void)n_in; (void)out_size; (void)ws_size;
  const float* x   = (const float*)d_in[0];
  const float* w1  = (const float*)d_in[1];
  const float* b1  = (const float*)d_in[2];
  const float* w2  = (const float*)d_in[3];
  const float* b2  = (const float*)d_in[4];
  const float* w3  = (const float*)d_in[5];
  const float* b3  = (const float*)d_in[6];
  const float* fw1 = (const float*)d_in[7];
  const float* fb1 = (const float*)d_in[8];
  const float* fw2 = (const float*)d_in[9];
  const float* fb2 = (const float*)d_in[10];
  float* out = (float*)d_out;

  char* ws = (char*)d_ws;
  // h1: [0, 44302336)  bf16 [B][13][13][16] -- dead after k_lc2, reused for h3/h4
  // h2: [44302336, 63176704)  bf16 [B][6][6][32]
  __bf16* h1 = (__bf16*)(ws);
  __bf16* h2 = (__bf16*)(ws + 44302336);
  __bf16* h3 = (__bf16*)(ws);                          // [B][1024] = 16777216 B
  __bf16* h4 = (__bf16*)(ws + 16777216);               // [B][512]  =  8388608 B

  k_lc1<<<dim3(5408),   dim3(256), 0, stream>>>(x,  w1,  b1,  h1);
  k_lc2<<<dim3(36, 64), dim3(256), 0, stream>>>(h1, w2,  b2,  h2);
  k_lc3<<<dim3(16, 64), dim3(256), 0, stream>>>(h2, w3,  b3,  h3);
  k_fc1<<<dim3(8, 64),  dim3(256), 0, stream>>>(h3, fw1, fb1, h4);
  k_fc2<<<dim3(320),    dim3(256), 0, stream>>>(h4, fw2, fb2, out);
}
